// ProtocolTreeAttention_52570399703524
// MI455X (gfx1250) — compile-verified, matrix operand order checked by reading, not run on hardware
//
#include <hip/hip_runtime.h>

// ProtocolTreeAttention for MI455X (gfx1250, wave32).
// All dense GEMMs via v_wmma_f32_16x16x32_f16 (f16 in, f32 accum).
// Hot kernels templated on compile-time dims so K-loops fully unroll:
// global loads pipeline over wmmas, hazard nops amortized.
// k_gemm: wave = 32x64 output block (8 accumulators, 8 wmma / 32-K step),
// block = 256 rows x 64-col strip; weight panel staged in LDS (8-wave reuse);
// epilogues staged through LDS for coalesced global_store_b128.

typedef _Float16 h16;
typedef __attribute__((ext_vector_type(16))) _Float16 v16h;
typedef __attribute__((ext_vector_type(8)))  _Float16 v8h;
typedef __attribute__((ext_vector_type(8)))  float    v8f;

#define HB 8192   // batch
#define NHEAD 4

// ---------------------------------------------------------------------------
// 16x32 f16 tile load matching CDNA5 WMMA A/B lane layout.
// lane<16: row=lane,  K in {0..7, 16..23}; lane>=16: row=lane-16, K in {8..15, 24..31}.
// Works for global or LDS pointers; rows must be 16B-aligned.
// ---------------------------------------------------------------------------
__device__ __forceinline__ v16h load_tile_f16(const h16* base, int ld, int lane) {
  const int r  = lane & 15;
  const int k0 = (lane & 16) ? 8 : 0;
  const h16* p = base + r * ld + k0;
  v8h lo = *(const v8h*)(p);
  v8h hi = *(const v8h*)(p + 16);
  v16h v;
#pragma unroll
  for (int j = 0; j < 8; ++j) { v[j] = lo[j]; v[8 + j] = hi[j]; }
  return v;
}

__device__ __forceinline__ float gelu_exact(float x) {
  return 0.5f * x * (1.0f + erff(x * 0.70710678118654752f));
}

// ---------------------------------------------------------------------------
// Weight prep: f32 -> f16, with/without [K,N] -> [N,K] transpose.
// ---------------------------------------------------------------------------
__global__ void k_tconv(const float* __restrict__ in, h16* __restrict__ out,
                        int G, int K, int N) {
  long idx = (long)blockIdx.x * blockDim.x + threadIdx.x;
  long total = (long)G * K * N;
  if (idx >= total) return;
  int n = (int)(idx % N);
  int k = (int)((idx / N) % K);
  int g = (int)(idx / ((long)N * K));
  out[((size_t)g * N + n) * K + k] = (h16)in[idx];
}

__global__ void k_conv(const float* __restrict__ in, h16* __restrict__ out, long n) {
  long i = (long)blockIdx.x * blockDim.x + threadIdx.x;
  if (i < n) out[i] = (h16)in[i];
}

// ---------------------------------------------------------------------------
// Per-field aligner: Y[b,f,:] = X[b,f,:32] @ W_f[32,E] + bias_f.  K=32 => one
// WMMA per 16x16 tile; wave covers 4 N-tiles (A reused 4x).  Compute pass
// (4 independent wmmas) separated from epilogue pass (hazard paid once).
// C staged in LDS then per-wave coalesced b128 stores.  Grid must be exact.
// ---------------------------------------------------------------------------
__global__ void k_align(const float* __restrict__ X, const h16* __restrict__ Wt,
                        const float* __restrict__ bias, h16* __restrict__ Y,
                        int F, int E) {
  __shared__ h16 cst[8 * 16 * 64];           // per-wave 16x64 staging
  const int lane   = threadIdx.x & 31;
  const int wavein = threadIdx.x >> 5;
  const int wave   = (int)(((long)blockIdx.x * blockDim.x + threadIdx.x) >> 5);
  const int ngrp = E >> 6;                   // 64-col groups
  const int ng = wave % ngrp;
  const int f  = (wave / ngrp) % F;
  const int mt = wave / (ngrp * F);

  const int r  = lane & 15;
  const int k0 = (lane & 16) ? 8 : 0;
  const float* xa = X + ((mt * 16 + r) * F + f) * 32 + k0;
  v16h a;
#pragma unroll
  for (int j = 0; j < 8; ++j) { a[j] = (h16)xa[j]; a[8 + j] = (h16)xa[16 + j]; }

  const h16* wbase = Wt + (f * E + ng * 64) * 32;
  v8f c[4];
#pragma unroll
  for (int t = 0; t < 4; ++t) {
    v16h b = load_tile_f16(wbase + t * 16 * 32, 32, lane);
    v8f z = {};
    c[t] = __builtin_amdgcn_wmma_f32_16x16x32_f16(false, a, false, b, (short)0, z, false, false);
  }

  const int m0 = (lane & 16) ? 8 : 0;
  const int nl = lane & 15;
  h16* my = cst + wavein * 16 * 64;
#pragma unroll
  for (int t = 0; t < 4; ++t) {
    const float bv = bias[f * E + ng * 64 + t * 16 + nl];
#pragma unroll
    for (int i = 0; i < 8; ++i)
      my[(m0 + i) * 64 + t * 16 + nl] = (h16)(c[t][i] + bv);
  }
  __syncthreads();
  // per-wave coalesced store: 16 rows x 64 halves = 128 v8h chunks
#pragma unroll
  for (int it = 0; it < 4; ++it) {
    const int chunk = it * 32 + lane;
    const int row  = chunk >> 3;
    const int col8 = (chunk & 7) * 8;
    v8h val = *(const v8h*)(my + row * 64 + col8);
    *(v8h*)(Y + ((mt * 16 + row) * F + f) * E + ng * 64 + col8) = val;
  }
}

// ---------------------------------------------------------------------------
// Blocked GEMM (K, ACT compile-time): Y[M,N] = act(X[M,K] @ Wt[N,K]^T + bias).
// Block = 8 waves; wave = 2 M-tiles x 4 N-tiles (32x64), 8 wmma per 32-K step,
// K-loop fully unrolled.  Block covers 256 rows x one 64-col strip.  B panel
// (64xK) staged in LDS, C re-staged after a barrier, block-cooperative b128
// stores.  Requires: M % 256 == 0, N % 64 == 0.  Launch with 32 KB LDS.
// ---------------------------------------------------------------------------
template <int K, int ACT>
__global__ void k_gemm(const h16* __restrict__ X, int ldX,
                       const h16* __restrict__ Wt,
                       const float* __restrict__ bias,
                       h16* __restrict__ Y, int ldY, int N) {
  extern __shared__ v8h lds_v[];             // phase1: B 64xK; phase2: C 256x64
  h16* lds_h = (h16*)lds_v;
  const int lane = threadIdx.x & 31;
  const int wave = threadIdx.x >> 5;
  const int nstrips = N >> 6;
  const int bm = blockIdx.x / nstrips;
  const int ns = blockIdx.x % nstrips;

  // stage B panel [64][K] into LDS
  const h16* wpanel = Wt + ns * 64 * K;
#pragma unroll
  for (int it = 0; it < (64 * K) / (8 * 256); ++it)
    lds_v[it * 256 + threadIdx.x] = ((const v8h*)wpanel)[it * 256 + threadIdx.x];
  __syncthreads();

  v8f c[2][4] = {};
  const int row0 = bm * 256 + wave * 32;
#pragma unroll
  for (int k = 0; k < K; k += 32) {
    v16h a0 = load_tile_f16(X + row0 * ldX + k, ldX, lane);
    v16h a1 = load_tile_f16(X + (row0 + 16) * ldX + k, ldX, lane);
#pragma unroll
    for (int t = 0; t < 4; ++t) {
      v16h b = load_tile_f16(lds_h + t * 16 * K + k, K, lane);
      c[0][t] = __builtin_amdgcn_wmma_f32_16x16x32_f16(false, a0, false, b, (short)0, c[0][t], false, false);
      c[1][t] = __builtin_amdgcn_wmma_f32_16x16x32_f16(false, a1, false, b, (short)0, c[1][t], false, false);
    }
  }
  __syncthreads();                           // done with B panel

  // bias + act, stage f16 C into LDS ordered by block row
  const int m0 = (lane & 16) ? 8 : 0;
  const int nl = lane & 15;
#pragma unroll
  for (int t = 0; t < 4; ++t) {
    const float bv = bias[ns * 64 + t * 16 + nl];
#pragma unroll
    for (int tm = 0; tm < 2; ++tm)
#pragma unroll
      for (int i = 0; i < 8; ++i) {
        float v = c[tm][t][i] + bv;
        if (ACT) v = gelu_exact(v);
        lds_h[(wave * 32 + tm * 16 + m0 + i) * 64 + t * 16 + nl] = (h16)v;
      }
  }
  __syncthreads();

  // block-cooperative coalesced store: 256 rows x 64 halves = 2048 v8h chunks
#pragma unroll
  for (int it = 0; it < 8; ++it) {
    const int chunk = it * 256 + threadIdx.x;
    const int rr   = chunk >> 3;
    const int col8 = (chunk & 7) * 8;
    v8h val = lds_v[chunk];
    *(v8h*)(Y + (bm * 256 + rr) * ldY + ns * 64 + col8) = val;
  }
}

// ---------------------------------------------------------------------------
// GEMM + residual + LayerNorm fused (E, K compile-time):
// H = LN(R + X@Wt^T + bias) * g + b.  One block = one 16-row M-tile x full E
// (waves = E/16).  Accumulators to LDS, wave32 __shfl_xor reductions; each
// lane owns a contiguous E/32 column slice so loads/stores pack.
// ---------------------------------------------------------------------------
template <int E, int K>
__global__ void k_gemm_res_ln(const h16* __restrict__ X, int ldX,
                              const h16* __restrict__ Wt,
                              const float* __restrict__ bias,
                              const h16* __restrict__ R, int ldR,
                              const float* __restrict__ lng,
                              const float* __restrict__ lnb,
                              h16* __restrict__ Hout, int ldH) {
  __shared__ float smem[16 * E];
  const int lane = threadIdx.x & 31;
  const int wave = threadIdx.x >> 5;
  constexpr int wpb = E >> 4;
  const int mt = blockIdx.x;
  const int nt = wave;

  v8f c = {};
  const h16* xrow = X + (mt * 16) * ldX;
  const h16* wrow = Wt + (nt * 16) * K;
#pragma unroll
  for (int k = 0; k < K; k += 32) {
    v16h a = load_tile_f16(xrow + k, ldX, lane);
    v16h b = load_tile_f16(wrow + k, K, lane);
    c = __builtin_amdgcn_wmma_f32_16x16x32_f16(false, a, false, b, (short)0, c, false, false);
  }
  const int n  = nt * 16 + (lane & 15);
  const int m0 = (lane & 16) ? 8 : 0;
  const float bv = bias[n];
#pragma unroll
  for (int i = 0; i < 8; ++i) {
    const int mr = m0 + i;
    const float rv = (float)R[(mt * 16 + mr) * ldR + n];
    smem[mr * E + n] = c[i] + bv + rv;
  }
  __syncthreads();
  constexpr int cpl = E >> 5;               // contiguous cols per lane (2 or 4)
#pragma unroll
  for (int row = wave; row < 16; row += wpb) {
    const float* rp = smem + row * E + lane * cpl;
    float s = 0.f, s2 = 0.f;
#pragma unroll
    for (int j = 0; j < cpl; ++j) { float x = rp[j]; s += x; s2 += x * x; }
#pragma unroll
    for (int off = 16; off > 0; off >>= 1) {
      s  += __shfl_xor(s, off);
      s2 += __shfl_xor(s2, off);
    }
    const float mean = s / E;
    const float inv  = rsqrtf(s2 / E - mean * mean + 1e-5f);
    h16* hp = Hout + (mt * 16 + row) * ldH + lane * cpl;
    const float* gp = lng + lane * cpl;
    const float* bp = lnb + lane * cpl;
#pragma unroll
    for (int j = 0; j < cpl; ++j)
      hp[j] = (h16)((rp[j] - mean) * inv * gp[j] + bp[j]);
  }
}

// ---------------------------------------------------------------------------
// Attention core (S, D compile-time): one thread per (b, head, q).
// Fully unrolled; scores stay in registers.  <5% of FLOPs.
// qkv layout [B,S,3E] f16 (q | k | v per token); O layout [B,S,E] f16.
// ---------------------------------------------------------------------------
template <int S, int D>
__global__ void k_attn(const h16* __restrict__ qkv, h16* __restrict__ O) {
  constexpr int E = D * NHEAD;
  const long total = (long)HB * NHEAD * S;
  const long idx = (long)blockIdx.x * blockDim.x + threadIdx.x;
  if (idx >= total) return;
  const int q  = (int)(idx % S);
  const int hh = (int)((idx / S) % NHEAD);
  const int b  = (int)(idx / (S * NHEAD));
  const h16* base = qkv + (size_t)b * S * 3 * E;
  const h16* qp = base + q * 3 * E + hh * D;
  const float scale = rsqrtf((float)D);
  float sc[S];
  float mx = -1e30f;
#pragma unroll
  for (int kk = 0; kk < S; ++kk) {
    const h16* kp = base + kk * 3 * E + E + hh * D;
    float acc = 0.f;
#pragma unroll
    for (int j = 0; j < D; ++j) acc += (float)qp[j] * (float)kp[j];
    acc *= scale;
    sc[kk] = acc;
    mx = fmaxf(mx, acc);
  }
  float den = 0.f;
#pragma unroll
  for (int kk = 0; kk < S; ++kk) { sc[kk] = __expf(sc[kk] - mx); den += sc[kk]; }
  const float dinv = 1.f / den;
  h16* op = O + ((size_t)b * S + q) * E + hh * D;
#pragma unroll
  for (int j = 0; j < D; ++j) {
    float acc = 0.f;
#pragma unroll
    for (int kk = 0; kk < S; ++kk)
      acc += sc[kk] * (float)base[kk * 3 * E + 2 * E + hh * D + j];
    op[j] = (h16)(acc * dinv);
  }
}

// ---------------------------------------------------------------------------
// Build xc = [CLS ; seg0 rows ; seg1 rows] as f16 [B,S,E].
// ---------------------------------------------------------------------------
__global__ void k_build_seq(h16* __restrict__ xc, int S, int E,
                            const float* __restrict__ cls,
                            const h16* __restrict__ seg0, long bs0, long rs0, int len0,
                            const h16* __restrict__ seg1, long bs1, long rs1) {
  long idx = (long)blockIdx.x * blockDim.x + threadIdx.x;
  long total = (long)HB * S * E;
  if (idx >= total) return;
  int e = (int)(idx % E);
  int s = (int)((idx / E) % S);
  long b = idx / ((long)E * S);
  float v;
  if (s == 0)            v = cls[e];
  else if (s - 1 < len0) v = (float)seg0[b * bs0 + (long)(s - 1) * rs0 + e];
  else                   v = (float)seg1[b * bs1 + (long)(s - 1 - len0) * rs1 + e];
  xc[((size_t)b * S + s) * E + e] = (h16)v;
}

__global__ void k_extract_cls(const h16* __restrict__ Hin, int S, int E,
                              h16* __restrict__ dst, long dstride) {
  long idx = (long)blockIdx.x * blockDim.x + threadIdx.x;
  long total = (long)HB * E;
  if (idx >= total) return;
  int e = (int)(idx % E);
  long b = idx / E;
  dst[b * dstride + e] = Hin[(size_t)b * S * E + e];
}

__global__ void k_clf(const h16* __restrict__ x, const h16* __restrict__ Wt,
                      const float* __restrict__ bias, float* __restrict__ out) {
  int idx = blockIdx.x * blockDim.x + threadIdx.x;
  if (idx >= HB * 2) return;
  int j = idx & 1;
  int b = idx >> 1;
  float acc = bias[j];
#pragma unroll 16
  for (int k = 0; k < 128; ++k)
    acc += (float)x[b * 128 + k] * (float)Wt[j * 128 + k];
  out[idx] = acc;
}

// ---------------------------------------------------------------------------
// Host driver
// ---------------------------------------------------------------------------
extern "C" void kernel_launch(void* const* d_in, const int* in_sizes, int n_in,
                              void* d_out, int out_size, void* d_ws, size_t ws_size,
                              hipStream_t stream) {
  (void)in_sizes; (void)n_in; (void)out_size; (void)ws_size;

  const float* leaf_feats   = (const float*)d_in[0];
  const float* sub_feats    = (const float*)d_in[1];
  const float* sub_align_W  = (const float*)d_in[2];
  const float* sub_align_b  = (const float*)d_in[3];
  const float* leaf_align_W = (const float*)d_in[4];
  const float* leaf_align_b = (const float*)d_in[5];
  const float* parent_align_W = (const float*)d_in[6];
  const float* parent_align_b = (const float*)d_in[7];
  const float* sA_cls  = (const float*)d_in[8];
  const float* sA_Wqkv = (const float*)d_in[9];
  const float* sA_bqkv = (const float*)d_in[10];
  const float* sA_Wo   = (const float*)d_in[11];
  const float* sA_bo   = (const float*)d_in[12];
  const float* sA_lng  = (const float*)d_in[13];
  const float* sA_lnb  = (const float*)d_in[14];
  const float* sA_W1   = (const float*)d_in[15];
  const float* sA_b1   = (const float*)d_in[16];
  const float* sA_W2   = (const float*)d_in[17];
  const float* sA_b2   = (const float*)d_in[18];
  const float* pA_cls  = (const float*)d_in[19];
  const float* pA_Wqkv = (const float*)d_in[20];
  const float* pA_bqkv = (const float*)d_in[21];
  const float* pA_Wo   = (const float*)d_in[22];
  const float* pA_bo   = (const float*)d_in[23];
  const float* pA_lng  = (const float*)d_in[24];
  const float* pA_lnb  = (const float*)d_in[25];
  const float* pA_W1   = (const float*)d_in[26];
  const float* pA_b1   = (const float*)d_in[27];
  const float* pA_W2   = (const float*)d_in[28];
  const float* pA_b2   = (const float*)d_in[29];
  const float* clf_W   = (const float*)d_in[30];
  const float* clf_b   = (const float*)d_in[31];

  h16* base = (h16*)d_ws;
  size_t off = 0;
  auto alloc = [&](size_t n) { h16* p = base + off; off += (n + 63) & ~(size_t)63; return p; };

  // f16 weights ([N][K] layout)
  h16* w_sub  = alloc((size_t)11 * 64 * 32);
  h16* w_leaf = alloc((size_t)34 * 128 * 32);
  h16* w_par  = alloc((size_t)2 * 128 * 64);
  h16* w_sqkv = alloc((size_t)2 * 192 * 64);
  h16* w_so   = alloc((size_t)2 * 64 * 64);
  h16* w_s1   = alloc((size_t)2 * 256 * 64);
  h16* w_s2   = alloc((size_t)2 * 64 * 256);
  h16* w_pqkv = alloc((size_t)4 * 384 * 128);
  h16* w_po   = alloc((size_t)4 * 128 * 128);
  h16* w_p1   = alloc((size_t)4 * 512 * 128);
  h16* w_p2   = alloc((size_t)4 * 128 * 512);
  h16* w_clf  = alloc((size_t)2 * 128);
  // activations (sized for the largest aggregator: S=17, E=128)
  h16* sub_al  = alloc((size_t)HB * 11 * 64);
  h16* leaf_al = alloc((size_t)HB * 34 * 128);
  h16* xc      = alloc((size_t)HB * 17 * 128);
  h16* big     = alloc((size_t)HB * 17 * 512);   // qkv (3E) then FFN hidden (4E)
  h16* obuf    = alloc((size_t)HB * 17 * 128);   // attn out, then final LN out
  h16* hbuf    = alloc((size_t)HB * 17 * 128);   // post-first-LN
  h16* tcpfv   = alloc((size_t)HB * 64);
  h16* ipfv    = alloc((size_t)HB * 64);
  h16* p_tcp   = alloc((size_t)HB * 128);
  h16* p_ip    = alloc((size_t)HB * 128);
  h16* vecs3   = alloc((size_t)HB * 3 * 128);    // [eth, ip, tcp]
  h16* pktv    = alloc((size_t)HB * 128);

  auto grid1 = [](long n, int bs) { return (unsigned)((n + bs - 1) / bs); };
  const unsigned GSMEM = 32768;   // k_gemm dynamic LDS (B panel + C staging)

  // ---- weight conversion (torch in_proj layouts [3E,E]/[E,E] are already [N][K]) ----
  k_tconv<<<grid1(11L*32*64,256),256,0,stream>>>(sub_align_W, w_sub, 11, 32, 64);
  k_tconv<<<grid1(34L*32*128,256),256,0,stream>>>(leaf_align_W, w_leaf, 34, 32, 128);
  k_tconv<<<grid1(2L*64*128,256),256,0,stream>>>(parent_align_W, w_par, 2, 64, 128);
  k_conv <<<grid1(2L*192*64,256),256,0,stream>>>(sA_Wqkv, w_sqkv, 2L*192*64);
  k_conv <<<grid1(2L*64*64,256),256,0,stream>>>(sA_Wo, w_so, 2L*64*64);
  k_tconv<<<grid1(2L*64*256,256),256,0,stream>>>(sA_W1, w_s1, 2, 64, 256);
  k_tconv<<<grid1(2L*256*64,256),256,0,stream>>>(sA_W2, w_s2, 2, 256, 64);
  k_conv <<<grid1(4L*384*128,256),256,0,stream>>>(pA_Wqkv, w_pqkv, 4L*384*128);
  k_conv <<<grid1(4L*128*128,256),256,0,stream>>>(pA_Wo, w_po, 4L*128*128);
  k_tconv<<<grid1(4L*128*512,256),256,0,stream>>>(pA_W1, w_p1, 4, 128, 512);
  k_tconv<<<grid1(4L*512*128,256),256,0,stream>>>(pA_W2, w_p2, 4, 512, 128);
  k_tconv<<<grid1(128L*2,256),256,0,stream>>>(clf_W, w_clf, 1, 128, 2);

  // ---- per-field aligners (K=32 WMMA, 4 tiles/wave); exact grids ----
  k_align<<<(HB/16)*11*1/8, 256, 0, stream>>>(sub_feats, w_sub, sub_align_b, sub_al, 11, 64);
  k_align<<<(HB/16)*34*2/8, 256, 0, stream>>>(leaf_feats, w_leaf, leaf_align_b, leaf_al, 34, 128);

  // ---- one attention-aggregator instance (S, E dispatched to templates) ----
  auto run_agg = [&](int S, int E,
                     const float* cls, const h16* Wqkv, const float* bqkv,
                     const h16* Wo, const float* bo,
                     const float* lng, const float* lnb,
                     const h16* W1t, const float* b1,
                     const h16* W2t, const float* b2,
                     const h16* seg0, long bs0, long rs0, int len0,
                     const h16* seg1, long bs1, long rs1,
                     h16* cls_out, long cls_stride) {
    const int M = HB * S;
    k_build_seq<<<grid1((long)HB*S*E,256),256,0,stream>>>(xc, S, E, cls,
        seg0, bs0, rs0, len0, seg1, bs1, rs1);
    // qkv projection: [M,E] @ [E,3E]
    if (E == 64) k_gemm<64,0> <<<(M/256)*(3*64/64), 256, GSMEM, stream>>>(xc, 64,  Wqkv, bqkv, big, 192, 192);
    else         k_gemm<128,0><<<(M/256)*(3*128/64),256, GSMEM, stream>>>(xc, 128, Wqkv, bqkv, big, 384, 384);
    // attention
    {
      const unsigned ag = grid1((long)HB*NHEAD*S, 256);
      if (E == 64) {
        if (S == 9) k_attn<9,16><<<ag,256,0,stream>>>(big, obuf);
        else        k_attn<4,16><<<ag,256,0,stream>>>(big, obuf);
      } else {
        if (S == 9)       k_attn<9,32> <<<ag,256,0,stream>>>(big, obuf);
        else if (S == 13) k_attn<13,32><<<ag,256,0,stream>>>(big, obuf);
        else if (S == 17) k_attn<17,32><<<ag,256,0,stream>>>(big, obuf);
        else              k_attn<4,32> <<<ag,256,0,stream>>>(big, obuf);
      }
    }
    // out-proj + residual(xc) + LN -> hbuf
    if (E == 64) k_gemm_res_ln<64,64>  <<<M/16, 128, 0, stream>>>(obuf, 64,  Wo, bo, xc, 64,  lng, lnb, hbuf, 64);
    else         k_gemm_res_ln<128,128><<<M/16, 256, 0, stream>>>(obuf, 128, Wo, bo, xc, 128, lng, lnb, hbuf, 128);
    // FFN up + exact GELU: [M,E] @ [E,4E]
    if (E == 64) k_gemm<64,1> <<<(M/256)*(4*64/64), 256, GSMEM, stream>>>(hbuf, 64,  W1t, b1, big, 256, 256);
    else         k_gemm<128,1><<<(M/256)*(4*128/64),256, GSMEM, stream>>>(hbuf, 128, W1t, b1, big, 512, 512);
    // FFN down + residual(hbuf) + LN -> obuf (reused as final h)
    if (E == 64) k_gemm_res_ln<64,256> <<<M/16, 128, 0, stream>>>(big, 256, W2t, b2, hbuf, 64,  lng, lnb, obuf, 64);
    else         k_gemm_res_ln<128,512><<<M/16, 256, 0, stream>>>(big, 512, W2t, b2, hbuf, 128, lng, lnb, obuf, 128);
    k_extract_cls<<<grid1((long)HB*E,256),256,0,stream>>>(obuf, S, E, cls_out, cls_stride);
  };

  // ---- subfield aggregators (E=64) ----
  run_agg(9, 64, sA_cls, w_sqkv, sA_bqkv, w_so, sA_bo, sA_lng, sA_lnb,
          w_s1, sA_b1, w_s2, sA_b2,
          sub_al, 11*64, 64, 8, sub_al, 11*64, 64, tcpfv, 64);                 // tcp.flags
  run_agg(4, 64, sA_cls+64, w_sqkv+192*64, sA_bqkv+192, w_so+64*64, sA_bo+64,
          sA_lng+64, sA_lnb+64, w_s1+256*64, sA_b1+256, w_s2+64*256, sA_b2+64,
          sub_al+8*64, 11*64, 64, 3, sub_al, 11*64, 64, ipfv, 64);             // ip.flags

  // ---- parent aligners 64 -> 128 ----
  k_gemm<64,0><<<(HB/256)*(128/64), 256, GSMEM, stream>>>(tcpfv, 64, w_par,        parent_align_b,     p_tcp, 128, 128);
  k_gemm<64,0><<<(HB/256)*(128/64), 256, GSMEM, stream>>>(ipfv,  64, w_par+128*64, parent_align_b+128, p_ip,  128, 128);

  // ---- field-group aggregators (E=128) ----
  run_agg(9, 128, pA_cls, w_pqkv, pA_bqkv, w_po, pA_bo, pA_lng, pA_lnb,
          w_p1, pA_b1, w_p2, pA_b2,
          leaf_al, 34*128, 128, 8, leaf_al, 34*128, 128, vecs3, 384);          // eth
  run_agg(13, 128, pA_cls+128, w_pqkv+384*128, pA_bqkv+384, w_po+128*128, pA_bo+128,
          pA_lng+128, pA_lnb+128, w_p1+512*128, pA_b1+512, w_p2+128*512, pA_b2+128,
          leaf_al+8*128, 34*128, 128, 11, p_ip, 128, 128, vecs3+128, 384);     // ip (+parent ip)
  run_agg(17, 128, pA_cls+2*128, w_pqkv+2*384*128, pA_bqkv+2*384, w_po+2*128*128, pA_bo+2*128,
          pA_lng+2*128, pA_lnb+2*128, w_p1+2*512*128, pA_b1+2*512, w_p2+2*128*512, pA_b2+2*128,
          leaf_al+19*128, 34*128, 128, 15, p_tcp, 128, 128, vecs3+256, 384);   // tcp (+parent tcp)
  run_agg(4, 128, pA_cls+3*128, w_pqkv+3*384*128, pA_bqkv+3*384, w_po+3*128*128, pA_bo+3*128,
          pA_lng+3*128, pA_lnb+3*128, w_p1+3*512*128, pA_b1+3*512, w_p2+3*128*512, pA_b2+3*128,
          vecs3, 3*128, 128, 3, vecs3, 3*128, 128, pktv, 128);                 // pkt

  // ---- classifier ----
  k_clf<<<grid1((long)HB*2,256),256,0,stream>>>(pktv, w_clf, clf_b, (float*)d_out);
}
